// ScaledDotProductAttention_64742337020450
// MI455X (gfx1250) — matrix operand order repaired and checked by reading, not compiled, not run on hardware
//
#include <hip/hip_runtime.h>

typedef __attribute__((ext_vector_type(16))) _Float16 v16h;
typedef __attribute__((ext_vector_type(8)))  float    v8f;

#define BATCH 4
#define HEADS 16
#define SEQ   2048
#define DIM   64
#define QTILE 128     // q rows per block (16 per wave, 8 waves)
#define KTILE 64      // keys per softmax step
#define KPAD  8
#define VPAD  8

__global__ __launch_bounds__(256)
void fa_fwd_causal_64742337020450(const float* __restrict__ Qg,
                                  const float* __restrict__ Kg,
                                  const float* __restrict__ Vg,
                                  float* __restrict__ Og)
{
    __shared__ _Float16 sK [KTILE][DIM   + KPAD];  // K tile, (key, d)
    __shared__ _Float16 sVt[DIM ][KTILE + VPAD];   // V tile transposed, (d, key)

    const int nqt  = SEQ / QTILE;
    const int bh   = blockIdx.x / nqt;
    const int qt   = blockIdx.x % nqt;

    const int tid  = threadIdx.x;
    const int wave = tid >> 5;
    const int lane = tid & 31;
    const int half = lane >> 4;   // 0: lanes 0-15, 1: lanes 16-31
    const int lr   = lane & 15;

    const size_t base  = (size_t)bh * SEQ * DIM;
    const int    qbase = qt * QTILE + wave * 16;
    const int    qm    = qbase + lr;          // this lane's query row (N-domain)

    // ---- Q^T as WMMA B-operand, softmax scale (1/8) folded in ----
    // B-layout: lane -> query column (lr); element i -> d = i + 16*half (+32 per hh)
    v16h qb[2];
    {
        const float* qp = Qg + base + (size_t)qm * DIM;
        #pragma unroll
        for (int hh = 0; hh < 2; ++hh) {
            #pragma unroll
            for (int i = 0; i < 16; ++i)
                qb[hh][i] = (_Float16)(qp[i + 16 * half + 32 * hh] * 0.125f);
        }
    }

    v16h onesb;                   // B-operand of all-ones: l += P @ 1 (row sums)
    #pragma unroll
    for (int i = 0; i < 16; ++i) onesb[i] = (_Float16)1.0f;

    v8f acc[4] = {};              // context, element r -> query qbase+r+8*half
    v8f lacc   = {};              // softmax denominators, same element domain
    float m_i  = -1.0e30f;        // running max for this lane's query

    const int njt = 2 * qt + 2;   // causal: 64-key tiles with jbase <= block qmax

    const int ldr = tid >> 2;            // 0..63 : tile row loaded by this thread
    const int ldc = (tid & 3) * 16;      // 0,16,32,48 : column base

    for (int jt = 0; jt < njt; ++jt) {
        const int jbase = jt * KTILE;

        // ---- cooperative K/V tile load (64x64), f32 -> f16, V transposed ----
        {
            const float* kp = Kg + base + (size_t)(jbase + ldr) * DIM + ldc;
            const float* vp = Vg + base + (size_t)(jbase + ldr) * DIM + ldc;
            #pragma unroll
            for (int u = 0; u < 4; ++u) {
                float4 kv = *(const float4*)(kp + u * 4);
                float4 vv = *(const float4*)(vp + u * 4);
                sK[ldr][ldc + u*4 + 0] = (_Float16)kv.x;
                sK[ldr][ldc + u*4 + 1] = (_Float16)kv.y;
                sK[ldr][ldc + u*4 + 2] = (_Float16)kv.z;
                sK[ldr][ldc + u*4 + 3] = (_Float16)kv.w;
                sVt[ldc + u*4 + 0][ldr] = (_Float16)vv.x;
                sVt[ldc + u*4 + 1][ldr] = (_Float16)vv.y;
                sVt[ldc + u*4 + 2][ldr] = (_Float16)vv.z;
                sVt[ldc + u*4 + 3][ldr] = (_Float16)vv.w;
            }
        }
        __syncthreads();

        // waves for which this whole tile is masked skip compute (wave-uniform)
        if (jbase <= qbase + 15) {
            // ---- S^T = K @ (Q^T/8); four 16x16 C tiles over key groups ----
            float s[32];   // s[g*8+r]: key = jbase + g*16 + r + 8*half, query = qm
            #pragma unroll
            for (int g = 0; g < 4; ++g) {
                v8f c = {};
                #pragma unroll
                for (int hh = 0; hh < 2; ++hh) {
                    v16h ka;   // A-layout: lane -> key row, elem i -> d
                    #pragma unroll
                    for (int i = 0; i < 16; ++i)
                        ka[i] = sK[g * 16 + lr][i + 16 * half + 32 * hh];
                    c = __builtin_amdgcn_wmma_f32_16x16x32_f16(false, ka, false, qb[hh],
                                                               (short)0, c, false, false);
                }
                #pragma unroll
                for (int r = 0; r < 8; ++r) s[g * 8 + r] = c[r];
            }

            // ---- causal mask only on diagonal-straddling tiles (wave-uniform) ----
            if (jbase + KTILE > qbase) {
                #pragma unroll
                for (int g = 0; g < 4; ++g)
                    #pragma unroll
                    for (int r = 0; r < 8; ++r) {
                        int kn = jbase + g * 16 + r + 8 * half;
                        if (kn > qm) s[g * 8 + r] = -1.0e9f;
                    }
            }

            // ---- online softmax: 31 in-lane maxes + one cross-half shuffle ----
            float mx = s[0];
            #pragma unroll
            for (int e = 1; e < 32; ++e) mx = fmaxf(mx, s[e]);
            mx = fmaxf(mx, __shfl_xor(mx, 16, 32));
            float mn = fmaxf(m_i, mx);
            float al = __expf(m_i - mn);
            m_i = mn;

            // probs straight into the A-operands of P@V (flat s[] ordering):
            // pa0 elem i -> key (i<8 ? i : i+8)+8*half (keys 0..31), pa1 -> +32
            v16h pa0, pa1;
            #pragma unroll
            for (int e = 0; e < 16; ++e) {
                pa0[e] = (_Float16)__expf(s[e]      - mn);
                pa1[e] = (_Float16)__expf(s[16 + e] - mn);
            }

            // ---- rescale accumulators: alpha lane-domain -> element-domain ----
            float ar[8];
            #pragma unroll
            for (int r = 0; r < 8; ++r) ar[r] = __shfl(al, r + 8 * half, 32);
            #pragma unroll
            for (int r = 0; r < 8; ++r) {
                acc[0][r] *= ar[r]; acc[1][r] *= ar[r];
                acc[2][r] *= ar[r]; acc[3][r] *= ar[r];
                lacc[r]   *= ar[r];
            }

            // ---- l += P @ 1 ; context += P @ V  (K-dim 64 = 2 WMMAs each) ----
            lacc = __builtin_amdgcn_wmma_f32_16x16x32_f16(false, pa0, false, onesb,
                                                          (short)0, lacc, false, false);
            lacc = __builtin_amdgcn_wmma_f32_16x16x32_f16(false, pa1, false, onesb,
                                                          (short)0, lacc, false, false);
            #pragma unroll
            for (int c = 0; c < 4; ++c) {
                v16h vb0, vb1;  // B-layout: lane -> d column, elem i -> key
                #pragma unroll
                for (int i = 0; i < 16; ++i) {
                    vb0[i] = sVt[c * 16 + lr][((i < 8) ? i : i + 8) + 8 * half];
                    vb1[i] = sVt[c * 16 + lr][32 + ((i < 8) ? i : i + 8) + 8 * half];
                }
                acc[c] = __builtin_amdgcn_wmma_f32_16x16x32_f16(false, pa0, false, vb0,
                                                                (short)0, acc[c], false, false);
                acc[c] = __builtin_amdgcn_wmma_f32_16x16x32_f16(false, pa1, false, vb1,
                                                                (short)0, acc[c], false, false);
            }
        }

        __syncthreads();   // protect sK/sVt before next tile's load
    }

    // ---- normalize and store fp32, coalesced per half-wave ----
    float* op = Og + base + (size_t)qbase * DIM;
    #pragma unroll
    for (int r = 0; r < 8; ++r) {
        float inv = 1.0f / lacc[r];
        #pragma unroll
        for (int c = 0; c < 4; ++c)
            op[(size_t)(r + 8 * half) * DIM + c * 16 + lr] = acc[c][r] * inv;
    }
}

extern "C" void kernel_launch(void* const* d_in, const int* in_sizes, int n_in,
                              void* d_out, int out_size, void* d_ws, size_t ws_size,
                              hipStream_t stream) {
    (void)in_sizes; (void)n_in; (void)d_ws; (void)ws_size; (void)out_size;
    const float* q = (const float*)d_in[0];
    const float* k = (const float*)d_in[1];
    const float* v = (const float*)d_in[2];
    // d_in[3] is the boolean causal mask: recomputed analytically in-kernel.
    float* o = (float*)d_out;

    dim3 grid(BATCH * HEADS * (SEQ / QTILE));   // 1024 blocks
    dim3 block(256);                            // 8 wave32 waves per block
    fa_fwd_causal_64742337020450<<<grid, block, 0, stream>>>(q, k, v, o);
}
